// WindowAttention3D_29265907155627
// MI455X (gfx1250) — matrix operand order repaired and checked by reading, not compiled, not run on hardware
//
#include <hip/hip_runtime.h>

// ---------------- problem constants (from reference) ----------------
#define B_WIN   4096
#define NTOK    98
#define CDIM    96
#define NHEADS  3
#define HDIM    32
#define NWIN    64
#define SCALE   0.17677669529663687f   // 1/sqrt(32)
#define SPSTR   128                    // padded K for P@V (4 k-steps of 32)
#define KTSTR   112                    // padded columns for K^T (7 N-tiles)
#define QKVW_T_ELEMS (CDIM * 3 * CDIM) // 96 x 288
#define PROJW_T_ELEMS (CDIM * CDIM)    // 96 x 96

typedef __attribute__((ext_vector_type(16))) __bf16         v16bf;
typedef __attribute__((ext_vector_type(16))) unsigned short v16u;
typedef __attribute__((ext_vector_type(8)))  unsigned short v8u;
typedef __attribute__((ext_vector_type(8)))  float          v8f;

__device__ __forceinline__ unsigned short f2bf(float f) {
  union { float f; unsigned u; } v; v.f = f;
  unsigned r = v.u + 0x7fffu + ((v.u >> 16) & 1u);   // round-to-nearest-even
  return (unsigned short)(r >> 16);
}
__device__ __forceinline__ float bf2f(unsigned short h) {
  union { unsigned u; float f; } v; v.u = ((unsigned)h) << 16;
  return v.f;
}

// A fragment: 16x32 bf16 from row-major [rows x stride] (LDS).
// ISA 7.12.2 layout: lane -> row m0+lane%16; elem i -> K = i + 8*(i>=8) + 8*(lane>=16).
// i=0..7 and i=8..15 are each 8 consecutive K -> two 16B ds_load_b128 per lane.
__device__ __forceinline__ v16bf load_a(const unsigned short* base, int stride,
                                        int m0, int k0, int mmax, int lane) {
  int m = m0 + (lane & 15); if (m > mmax) m = mmax;   // row clamp (dupes discarded)
  int ko = (lane & 16) ? 8 : 0;
  const v8u* p = (const v8u*)(base + m * stride + k0 + ko);
  v8u lo = p[0];   // K = k0+ko   .. k0+ko+7
  v8u hi = p[2];   // K = k0+ko+16.. k0+ko+23
  v16u u = __builtin_shufflevector(lo, hi, 0,1,2,3,4,5,6,7,8,9,10,11,12,13,14,15);
  return __builtin_bit_cast(v16bf, u);
}

// B fragment: 32x16 bf16 from row-major [K x stride] (LDS or global bf16).
// Layout: lane -> column n0+lane%16; elem i -> K = k0 + i + 8*(i>=8) + 8*(lane>=16) (clamped)
// -> strided scalar loads per lane.
__device__ __forceinline__ v16bf load_b(const unsigned short* base, int stride,
                                        int k0, int n0, int kmax, int lane) {
  int n = n0 + (lane & 15);
  int kb = k0 + ((lane & 16) ? 8 : 0);
  v16u u;
#pragma unroll
  for (int s = 0; s < 16; ++s) {
    int k = kb + s + ((s >= 8) ? 8 : 0); if (k > kmax) k = kmax;
    u[s] = base[k * stride + n];
  }
  return __builtin_bit_cast(v16bf, u);
}

// =====================================================================
// Kernel 0: one-shot weight pre-transpose + bf16 convert into workspace.
//   wTg[c*288 + j] = bf16(qkv_w[j][c]),  pTg[c*96 + j] = bf16(proj_w[j][c])
// =====================================================================
__global__ __launch_bounds__(256)
void cvt_weights(const float* __restrict__ qkv_w, const float* __restrict__ proj_w,
                 unsigned short* __restrict__ wTg, unsigned short* __restrict__ pTg) {
  int i = blockIdx.x * 256 + threadIdx.x;
  if (i < QKVW_T_ELEMS) {               // qkv_w is [288][96] row-major
    int j = i / CDIM, c = i % CDIM;
    wTg[c * (3 * CDIM) + j] = f2bf(qkv_w[i]);
  }
  if (i < PROJW_T_ELEMS) {              // proj_w is [96][96] row-major
    int j = i / CDIM, c = i % CDIM;
    pTg[c * CDIM + j] = f2bf(proj_w[i]);
  }
}

// =====================================================================
// Kernel 1: fused window attention (QKV -> S+bias+mask -> softmax -> PV)
// One block per window; 8 waves. y (pre-projection, bf16) -> workspace.
// =====================================================================
__global__ __launch_bounds__(256)
void win_attn_fused(const float* __restrict__ x,
                    const unsigned short* __restrict__ wTg,   // [96][288] bf16 (W^T)
                    const float* __restrict__ qkv_b,
                    const float* __restrict__ bias_table,
                    const float* __restrict__ attn_mask,
                    const int*   __restrict__ rel_idx,
                    unsigned short* __restrict__ y) {
  __shared__ alignas(16) unsigned short xs[NTOK * CDIM];    // 18816 B
  __shared__ alignas(16) unsigned short qs[NTOK * HDIM];    //  6272 B
  __shared__ alignas(16) unsigned short kT[HDIM * KTSTR];   //  7168 B (cols 98..111 zero)
  __shared__ alignas(16) unsigned short vs[NTOK * HDIM];    //  6272 B
  __shared__ alignas(16) unsigned short sp[NTOK * SPSTR];   // 25088 B
                                                            // total 63616 B

  const int tid  = threadIdx.x;
  const int lane = tid & 31;
  const int wv   = tid >> 5;
  const int b    = blockIdx.x;
  const int wi   = b & (NWIN - 1);

  // warm L2/WGP$ for the shared weight buffer (global_prefetch_b8)
  __builtin_prefetch(wTg + tid * 108, 0, 1);

  // zero pad regions once: sp cols 98..127 must stay 0 (P@V), kT cols 98..111 (S B-frags)
  for (int i = tid; i < NTOK * SPSTR; i += 256) sp[i] = 0;
  for (int i = tid; i < HDIM * KTSTR; i += 256) kT[i] = 0;

  // stage x window as bf16
  const float* xb = x + (size_t)b * NTOK * CDIM;
  for (int i = tid; i < NTOK * CDIM; i += 256) xs[i] = f2bf(xb[i]);
  __syncthreads();

  for (int h = 0; h < NHEADS; ++h) {
    // ---- QKV for this head: [98x96] @ W^T slice, 7x6 tiles, K=3 steps ----
    for (int t = wv; t < 42; t += 8) {
      int mt = t / 6, nt = t % 6;
      int m0 = mt * 16;
      int g  = nt >> 1;                  // 0=q 1=k 2=v
      int d0 = (nt & 1) * 16;
      int jbase = g * CDIM + h * HDIM + d0;
      v8f acc = {};
#pragma unroll
      for (int ks = 0; ks < 3; ++ks) {
        int k0 = ks * 32;
        v16bf a  = load_a(xs, CDIM, m0, k0, NTOK - 1, lane);
        v16bf bw = load_b(wTg, 3 * CDIM, k0, jbase, CDIM - 1, lane);
        acc = __builtin_amdgcn_wmma_f32_16x16x32_bf16(false, a, false, bw,
                                                      (short)0, acc, false, false);
      }
      int nl = lane & 15;
      float bias = qkv_b[jbase + nl];
#pragma unroll
      for (int r = 0; r < 8; ++r) {
        int mm = m0 + r + ((lane & 16) ? 8 : 0);
        if (mm < NTOK) {
          float v = acc[r] + bias;
          int d = d0 + nl;
          if (g == 0)      qs[mm * HDIM + d]  = f2bf(v * SCALE);
          else if (g == 1) kT[d * KTSTR + mm] = f2bf(v);
          else             vs[mm * HDIM + d]  = f2bf(v);
        }
      }
    }
    __syncthreads();

    // ---- S = Q @ K^T + rel_bias + mask : 7x7 tiles, one WMMA each (K=32) ----
    for (int t = wv; t < 49; t += 8) {
      int mt = t / 7, nt = t % 7;
      int m0 = mt * 16, n0 = nt * 16;
      v8f acc = {};
      v16bf a  = load_a(qs, HDIM, m0, 0, NTOK - 1, lane);
      v16bf bk = load_b(kT, KTSTR, 0, n0, HDIM - 1, lane);
      acc = __builtin_amdgcn_wmma_f32_16x16x32_bf16(false, a, false, bk,
                                                    (short)0, acc, false, false);
      int nn = n0 + (lane & 15);
#pragma unroll
      for (int r = 0; r < 8; ++r) {
        int mm = m0 + r + ((lane & 16) ? 8 : 0);
        if (mm < NTOK && nn < NTOK) {
          float s = acc[r]
                  + bias_table[rel_idx[mm * NTOK + nn] * NHEADS + h]
                  + attn_mask[((size_t)wi * NTOK + mm) * NTOK + nn];
          sp[mm * SPSTR + nn] = f2bf(s);
        }
      }
    }
    __syncthreads();

    // ---- row softmax (one row per thread; 98 rows) ----
    if (tid < NTOK) {
      unsigned short* row = sp + tid * SPSTR;
      float mx = -1e30f;
      for (int j = 0; j < NTOK; ++j) { float v = bf2f(row[j]); mx = fmaxf(mx, v); }
      float sum = 0.0f;
      for (int j = 0; j < NTOK; ++j) {
        float e = __expf(bf2f(row[j]) - mx);
        sum += e;
        row[j] = f2bf(e);
      }
      float inv = 1.0f / sum;
      for (int j = 0; j < NTOK; ++j) row[j] = f2bf(bf2f(row[j]) * inv);
    }
    __syncthreads();

    // ---- out_h = P @ V : 7x2 tiles, K=128 (P cols 98..127 are zero) ----
    for (int t = wv; t < 14; t += 8) {
      int mt = t >> 1, d0 = (t & 1) * 16;
      int m0 = mt * 16;
      v8f acc = {};
#pragma unroll
      for (int ks = 0; ks < 4; ++ks) {
        int k0 = ks * 32;
        v16bf a  = load_a(sp, SPSTR, m0, k0, NTOK - 1, lane);
        v16bf bv = load_b(vs, HDIM, k0, d0, NTOK - 1, lane);   // row-clamped, P=0 there
        acc = __builtin_amdgcn_wmma_f32_16x16x32_bf16(false, a, false, bv,
                                                      (short)0, acc, false, false);
      }
      int nl = lane & 15;
#pragma unroll
      for (int r = 0; r < 8; ++r) {
        int mm = m0 + r + ((lane & 16) ? 8 : 0);
        if (mm < NTOK)
          y[((size_t)b * NTOK + mm) * CDIM + h * HDIM + d0 + nl] = f2bf(acc[r]);
      }
    }
    __syncthreads();   // before next head reuses qs/kT/vs/sp
  }
}

// =====================================================================
// Kernel 2: projection  out = y @ proj_w^T + proj_b  (f32 out)
// 128 rows per block (4096*98 = 401408 = 3136*128 exactly).
// =====================================================================
__global__ __launch_bounds__(256)
void proj_gemm(const unsigned short* __restrict__ y,
               const unsigned short* __restrict__ pTg,  // [96][96] bf16 (W^T)
               const float* __restrict__ proj_b,
               float* __restrict__ out) {
  __shared__ alignas(16) unsigned short ys[128 * CDIM];  // 24576 B

  const int tid  = threadIdx.x;
  const int lane = tid & 31;
  const int wv   = tid >> 5;
  const size_t r0 = (size_t)blockIdx.x * 128;

  // 16B vector copy of the 128x96 activation tile
  {
    const v8u* src = (const v8u*)(y + r0 * CDIM);
    v8u* dst = (v8u*)ys;
    for (int i = tid; i < (128 * CDIM) / 8; i += 256) dst[i] = src[i];
  }
  __syncthreads();

  for (int t = wv; t < 48; t += 8) {          // 8 M-tiles x 6 N-tiles
    int mt = t / 6, nt = t % 6;
    int m0 = mt * 16, n0 = nt * 16;
    v8f acc = {};
#pragma unroll
    for (int ks = 0; ks < 3; ++ks) {
      int k0 = ks * 32;
      v16bf a  = load_a(ys, CDIM, m0, k0, 127, lane);
      v16bf bw = load_b(pTg, CDIM, k0, n0, CDIM - 1, lane);
      acc = __builtin_amdgcn_wmma_f32_16x16x32_bf16(false, a, false, bw,
                                                    (short)0, acc, false, false);
    }
    int nl = lane & 15;
    float bias = proj_b[n0 + nl];
#pragma unroll
    for (int r = 0; r < 8; ++r) {
      int mm = m0 + r + ((lane & 16) ? 8 : 0);
      out[(r0 + mm) * CDIM + n0 + nl] = acc[r] + bias;
    }
  }
}

// =====================================================================
extern "C" void kernel_launch(void* const* d_in, const int* in_sizes, int n_in,
                              void* d_out, int out_size, void* d_ws, size_t ws_size,
                              hipStream_t stream) {
  (void)in_sizes; (void)n_in; (void)out_size; (void)ws_size;
  const float* x          = (const float*)d_in[0];
  const float* qkv_w      = (const float*)d_in[1];
  const float* qkv_b      = (const float*)d_in[2];
  const float* proj_w     = (const float*)d_in[3];
  const float* proj_b     = (const float*)d_in[4];
  const float* bias_table = (const float*)d_in[5];
  const float* attn_mask  = (const float*)d_in[6];
  const int*   rel_idx    = (const int*)d_in[7];
  // d_in[8] = nW (hardcoded 64 to match setup_inputs)

  // workspace layout (all 16B-aligned)
  unsigned short* y   = (unsigned short*)d_ws;                    // 4096*98*96 bf16
  unsigned short* wTg = y + (size_t)B_WIN * NTOK * CDIM;          // 96*288 bf16
  unsigned short* pTg = wTg + QKVW_T_ELEMS;                       // 96*96  bf16
  float* out = (float*)d_out;

  cvt_weights<<<dim3((QKVW_T_ELEMS + 255) / 256), dim3(256), 0, stream>>>(
      qkv_w, proj_w, wTg, pTg);

  win_attn_fused<<<dim3(B_WIN), dim3(256), 0, stream>>>(
      x, wTg, qkv_b, bias_table, attn_mask, rel_idx, y);

  proj_gemm<<<dim3((B_WIN * NTOK) / 128), dim3(256), 0, stream>>>(
      y, pTg, proj_b, out);
}
